// CrossAttention_41910290874966
// MI455X (gfx1250) — matrix operand order repaired and checked by reading, not compiled, not run on hardware
//
#include <hip/hip_runtime.h>

typedef __attribute__((ext_vector_type(16))) _Float16 v16h;
typedef __attribute__((ext_vector_type(8)))  _Float16 v8h;
typedef __attribute__((ext_vector_type(8)))  float    v8f;
typedef __attribute__((ext_vector_type(4)))  unsigned int v4u;
typedef __attribute__((ext_vector_type(8)))  int      v8i;
typedef __attribute__((ext_vector_type(4)))  int      v4i;

#define B_ 4
#define X_ 32
#define H_ 8
#define W_ 128
#define C_ 32
#define N_ 4096   /* tokens per batch = X*W  */
#define K_ 256    /* features        = C*H  */
#define WAVES 4
#define NCHUNK (N_ / 32)

// ---------------------------------------------------------------------------
// Kernel 1: Q projection -> Q^T [b][k][n] (f16, feature-major).
// ---------------------------------------------------------------------------
__global__ void prep_q(const float* __restrict__ target,
                       const float* __restrict__ w_q,
                       const float* __restrict__ b_q,
                       _Float16* __restrict__ qt) {
    int tid = blockIdx.x * blockDim.x + threadIdx.x;   // B*C*W*H = 131072
    int h = tid & 7;
    int w = (tid >> 3) & 127;
    int c = (tid >> 10) & 31;
    int b = tid >> 15;
    const float* tv = target + (((size_t)b * H_ + h) * W_ + w) * C_;
    float acc = b_q[c];
#pragma unroll 8
    for (int ci = 0; ci < C_; ++ci) acc += w_q[c * C_ + ci] * tv[ci];
    _Float16 hv = (_Float16)acc;
    int kk    = (w & 31) * 8 + h;
    int nbase = c * 128 + (w >> 5);
    _Float16* row = qt + ((size_t)b * K_ + kk) * (size_t)N_ + nbase;
#pragma unroll
    for (int x = 0; x < X_; ++x) row[x * 4] = hv;      // broadcast over depth
}

// ---------------------------------------------------------------------------
// Kernel 2: Conv3d (3,1,1)/pad(1,0,0) -> K,V token-major [b][m][k] f16.
// ---------------------------------------------------------------------------
__global__ void prep_kv(const float* __restrict__ storage,
                        const float* __restrict__ w_cross,
                        const float* __restrict__ b_cross,
                        _Float16* __restrict__ km,
                        _Float16* __restrict__ vm) {
    int tid = blockIdx.x * blockDim.x + threadIdx.x;   // 4194304
    int h = tid & 7;
    int w = (tid >> 3) & 127;
    int x = (tid >> 10) & 31;
    int c = (tid >> 15) & 31;
    int b = tid >> 20;
    float accV = b_cross[c];
    float accK = b_cross[C_ + c];
#pragma unroll
    for (int d = 0; d < 3; ++d) {
        int xs = x + d - 1;
        if (xs < 0 || xs >= X_) continue;
        const float* sv = storage + ((((size_t)b * X_ + xs) * H_ + h) * W_ + w) * C_;
        const float* wv = w_cross + ((size_t)c * C_) * 3 + d;
        const float* wk = w_cross + ((size_t)(C_ + c) * C_) * 3 + d;
#pragma unroll 8
        for (int ci = 0; ci < C_; ++ci) {
            float s = sv[ci];
            accV += wv[ci * 3] * s;
            accK += wk[ci * 3] * s;
        }
    }
    int m  = c * 128 + x * 4 + (w >> 5);
    int kk = (w & 31) * 8 + h;
    size_t o = ((size_t)b * N_ + m) * K_ + kk;
    vm[o] = (_Float16)accV;
    km[o] = (_Float16)accK;
}

// ---------------------------------------------------------------------------
// TDM tile load: 2D tensor, 2-byte elements, row length K_=256, tile rows x 256.
// D# bitfields per CDNA5 ISA ch.8 (extra groups zero => dims 3..5 unused).
// This toolchain exposes the 6-arg builtin: (v4u, v8i, v4i, v4i, v8i, cpol).
// ---------------------------------------------------------------------------
#if __has_builtin(__builtin_amdgcn_tensor_load_to_lds)
__device__ __forceinline__ void tdm_load_tile(const _Float16* gsrc, unsigned lds_byte,
                                              int rows) {
    unsigned long long ga = (unsigned long long)(uintptr_t)gsrc;
    v4u g0;
    g0[0] = 1u;                                           // count=1, user descriptor
    g0[1] = lds_byte;                                     // lds_addr
    g0[2] = (unsigned)(ga & 0xffffffffu);                 // global_addr lo
    g0[3] = (unsigned)((ga >> 32) & 0x01ffffffu) | 0x80000000u; // addr hi | type=2
    v8i g1;
    g1[0] = 0x00010000;                                   // data_size=1 (2B), mask=0
    g1[1] = (int)(K_ << 16);                              // tensor_dim0 = 256
    g1[2] = (int)(N_ << 16);                              // tensor_dim1 = 4096
    g1[3] = (int)(K_ << 16);                              // tile_dim0 = 256
    g1[4] = rows;                                         // tile_dim1 = rows
    g1[5] = K_;                                           // tensor_dim0_stride = 256
    g1[6] = 0;
    g1[7] = 0;
    v4i z4 = {0, 0, 0, 0};
    v8i z8 = {0, 0, 0, 0, 0, 0, 0, 0};
    __builtin_amdgcn_tensor_load_to_lds(g0, g1, z4, z4, z8, 0);
}
#define STAGE_WAIT() { __builtin_amdgcn_s_wait_tensorcnt(0); asm volatile("" ::: "memory"); }
#define STAGE_IS_TDM 1
#else
// Fallback: per-lane async copies to LDS (ASYNCcnt-tracked).
#define STAGE_IS_TDM 0
#define STAGE_WAIT() { asm volatile("s_wait_asynccnt 0x0" ::: "memory"); }
#endif

// ---------------------------------------------------------------------------
// Kernel 3: flash attention. 4 waves/block share one batch; K/V chunks are
// TDM-streamed into a double-buffered LDS tile while WMMAs consume the
// previous chunk from LDS. One wave = 16 queries.
// ---------------------------------------------------------------------------
__global__ __launch_bounds__(WAVES * 32)
void attn(const _Float16* __restrict__ qt,
          const _Float16* __restrict__ km,
          const _Float16* __restrict__ vm,
          float* __restrict__ out) {
    __shared__ v16h     lds_q[WAVES * 8 * 32];        // 32 KB
    __shared__ _Float16 lds_kv[2 * 64 * K_];          // 2 x (32 K-rows + 32 V-rows)

    int lane   = threadIdx.x & 31;
    int wave   = threadIdx.x >> 5;
    int gw     = blockIdx.x * WAVES + wave;           // 0..1023
    int b      = gw >> 8;
    int tile   = gw & 255;
    int nb     = tile * 16;
    int lane16 = lane & 15;
    int half   = lane >> 4;

    const _Float16* kbase = km + (size_t)b * N_ * K_;
    const _Float16* vbase = vm + (size_t)b * N_ * K_;

    // Stage Q^T B-fragments once: lane <-> query nb+lane16, 16 f16 = k rows (table order).
#pragma unroll
    for (int kc = 0; kc < 8; ++kc) {
        v16h t;
#pragma unroll
        for (int s = 0; s < 16; ++s) {
            int kk = kc * 32 + (s < 8 ? half * 8 + s : 16 + half * 8 + (s - 8));
            t[s] = qt[((size_t)b * K_ + (size_t)kk) * N_ + nb + lane16];
        }
        lds_q[(wave * 8 + kc) * 32 + lane] = t;
    }

    v8f o[16];
#pragma unroll
    for (int kt = 0; kt < 16; ++kt) o[kt] = (v8f)0.0f;
    float mrun = -1e30f, lrun = 0.0f;

    // ---- issue DMA for chunk 0 ----
#if STAGE_IS_TDM
    if (wave == 0) {
        tdm_load_tile(kbase, (unsigned)(uintptr_t)&lds_kv[0], 32);
        tdm_load_tile(vbase, (unsigned)(uintptr_t)&lds_kv[32 * K_], 32);
    }
#else
    {
        int tid = threadIdx.x;
#pragma unroll
        for (int j = 0; j < 16; ++j) {                 // 2048 b128 pieces / 128 thr
            int idx = tid + j * 128;
            int row = idx >> 5, seg = idx & 31;        // row 0..63, 16B segment
            const _Float16* g = (row < 32 ? kbase + (size_t)row * K_
                                          : vbase + (size_t)(row - 32) * K_) + seg * 8;
            unsigned loff = (unsigned)(uintptr_t)&lds_kv[row * K_ + seg * 8];
            asm volatile("global_load_async_to_lds_b128 %0, %1, off"
                         :: "v"(loff), "v"(g) : "memory");
        }
    }
#endif

    for (int mc = 0; mc < NCHUNK; ++mc) {
        int cur = mc & 1;
#if STAGE_IS_TDM
        if (wave == 0) STAGE_WAIT();
#else
        STAGE_WAIT();
#endif
        __syncthreads();                               // buf[cur] ready & visible

        // ---- issue DMA for chunk mc+1 into the other buffer ----
        if (mc + 1 < NCHUNK) {
            int mb1 = (mc + 1) * 32, nxt = (mc + 1) & 1;
#if STAGE_IS_TDM
            if (wave == 0) {
                tdm_load_tile(kbase + (size_t)mb1 * K_,
                              (unsigned)(uintptr_t)&lds_kv[(size_t)nxt * 64 * K_], 32);
                tdm_load_tile(vbase + (size_t)mb1 * K_,
                              (unsigned)(uintptr_t)&lds_kv[(size_t)nxt * 64 * K_ + 32 * K_], 32);
            }
#else
            int tid = threadIdx.x;
#pragma unroll
            for (int j = 0; j < 16; ++j) {
                int idx = tid + j * 128;
                int row = idx >> 5, seg = idx & 31;
                const _Float16* g = (row < 32 ? kbase + (size_t)(mb1 + row) * K_
                                              : vbase + (size_t)(mb1 + row - 32) * K_) + seg * 8;
                unsigned loff =
                    (unsigned)(uintptr_t)&lds_kv[(size_t)nxt * 64 * K_ + row * K_ + seg * 8];
                asm volatile("global_load_async_to_lds_b128 %0, %1, off"
                             :: "v"(loff), "v"(g) : "memory");
            }
#endif
        }

        // ---- compute from LDS: S^T tiles (A = K rows, B = Q^T) ----
        const _Float16* kl    = lds_kv + (size_t)cur * 64 * K_;
        const _Float16* krow0 = kl + (size_t)lane16 * K_;          // tile0 row
        const _Float16* krow1 = krow0 + 16 * K_;                   // tile1 row
        v8f st0 = (v8f)0.0f, st1 = (v8f)0.0f;
#pragma unroll
        for (int kc = 0; kc < 8; ++kc) {
            v16h bq = lds_q[(wave * 8 + kc) * 32 + lane];
            v8h lo0 = *(const v8h*)(krow0 + kc * 32 + half * 8);
            v8h hi0 = *(const v8h*)(krow0 + kc * 32 + 16 + half * 8);
            v8h lo1 = *(const v8h*)(krow1 + kc * 32 + half * 8);
            v8h hi1 = *(const v8h*)(krow1 + kc * 32 + 16 + half * 8);
            v16h a0, a1;
#pragma unroll
            for (int i = 0; i < 8; ++i) {
                a0[i] = lo0[i]; a0[8 + i] = hi0[i];
                a1[i] = lo1[i]; a1[8 + i] = hi1[i];
            }
            st0 = __builtin_amdgcn_wmma_f32_16x16x32_f16(false, a0, false, bq,
                                                         (short)0, st0, false, false);
            st1 = __builtin_amdgcn_wmma_f32_16x16x32_f16(false, a1, false, bq,
                                                         (short)0, st1, false, false);
        }

        // ---- online softmax (lane <-> query n = lane16, vgprs <-> keys) ----
        float cm = -1e30f;
#pragma unroll
        for (int r = 0; r < 8; ++r) cm = fmaxf(cm, fmaxf(st0[r], st1[r]));
        cm = fmaxf(cm, __shfl_xor(cm, 16, 32));
        float newm = fmaxf(mrun, cm);
        if (__any(cm > mrun)) {                        // wave-uniform branch
            float scale = __expf(mrun - newm);
            float sb[8];
#pragma unroll
            for (int r = 0; r < 8; ++r) sb[r] = __shfl(scale, half * 8 + r, 32);
#pragma unroll
            for (int kt = 0; kt < 16; ++kt)
#pragma unroll
                for (int r = 0; r < 8; ++r) o[kt][r] *= sb[r];
            lrun *= scale;
        }
        mrun = newm;

        float ls = 0.0f;
        v16h pa;
#pragma unroll
        for (int r = 0; r < 8; ++r) {
            float p0 = __expf(st0[r] - mrun);
            float p1 = __expf(st1[r] - mrun);
            ls += p0 + p1;
            pa[r] = (_Float16)p0;                      // D-of-S^T == A-of-P layout
            pa[8 + r] = (_Float16)p1;
        }
        ls += __shfl_xor(ls, 16, 32);
        lrun += ls;

        // ---- O += P(16x32) . V(32x16 per feature tile), V from LDS ----
        const _Float16* vcol = kl + (size_t)32 * K_ + lane16;
#pragma unroll
        for (int kt = 0; kt < 16; ++kt) {
            v16h bv;
#pragma unroll
            for (int s = 0; s < 16; ++s) {
                int kk = (s < 8) ? half * 8 + s : 16 + half * 8 + (s - 8);
                bv[s] = vcol[(size_t)kk * K_ + kt * 16];
            }
            o[kt] = __builtin_amdgcn_wmma_f32_16x16x32_f16(false, pa, false, bv,
                                                           (short)0, o[kt], false, false);
        }
    }

    // ---- epilogue: divide by row sums, store reference-flat [b][n][k] ----
    float invl = 1.0f / lrun;
    float sb[8];
#pragma unroll
    for (int r = 0; r < 8; ++r) sb[r] = __shfl(invl, half * 8 + r, 32);
    float* ob = out + ((size_t)b * N_ + nb) * (size_t)K_;
#pragma unroll
    for (int kt = 0; kt < 16; ++kt)
#pragma unroll
        for (int r = 0; r < 8; ++r)
            ob[(size_t)(half * 8 + r) * K_ + kt * 16 + lane16] = o[kt][r] * sb[r];
}

// ---------------------------------------------------------------------------
extern "C" void kernel_launch(void* const* d_in, const int* in_sizes, int n_in,
                              void* d_out, int out_size, void* d_ws, size_t ws_size,
                              hipStream_t stream) {
    (void)in_sizes; (void)n_in; (void)out_size; (void)ws_size;
    const float* storage = (const float*)d_in[0];
    const float* target  = (const float*)d_in[1];
    const float* w_cross = (const float*)d_in[2];
    const float* b_cross = (const float*)d_in[3];
    const float* w_q     = (const float*)d_in[4];
    const float* b_q     = (const float*)d_in[5];

    _Float16* qt = (_Float16*)d_ws;                    // [B][256][4096]  8 MB
    _Float16* km = qt + (size_t)B_ * K_ * N_;          // [B][4096][256]  8 MB
    _Float16* vm = km + (size_t)B_ * N_ * K_;          // [B][4096][256]  8 MB

    prep_q <<<(B_ * C_ * W_ * H_) / 256, 256, 0, stream>>>(target, w_q, b_q, qt);
    prep_kv<<<(B_ * C_ * X_ * W_ * H_) / 256, 256, 0, stream>>>(storage, w_cross, b_cross, km, vm);
    attn   <<<(B_ * (N_ / 16)) / WAVES, WAVES * 32, 0, stream>>>(qt, km, vm, (float*)d_out);
}
